// E3nnVBnet_11192684774056
// MI455X (gfx1250) — matrix operand order repaired and checked by reading, not compiled, not run on hardware
//
#include <hip/hip_runtime.h>
#include <hip/hip_bf16.h>
#include <math.h>

typedef __attribute__((ext_vector_type(16))) _Float16 v16h;
typedef __attribute__((ext_vector_type(8)))  float    v8f;

#define N_CHC     16
#define HIDC      48
#define N_LAYERSC 4
#define N_GRAPHSC 64
#define TPB       256
#define EPB       256   // edges per block
#define TP_NORM   0.1767766952966369f  // 1/sqrt(2*16)

__device__ __forceinline__ float silu_f(float x) {
    return x / (1.0f + __expf(-x));
}

// ---------------------------------------------------------------- embed: h = x @ embed_w
__global__ void embed_kernel(const float* __restrict__ x, const float* __restrict__ w,
                             float* __restrict__ h, int n_nodes) {
    int idx = blockIdx.x * blockDim.x + threadIdx.x;
    if (idx >= n_nodes * HIDC) return;
    int node = idx / HIDC, ch = idx % HIDC;
    float acc = x[node * 3 + 0] * w[0 * HIDC + ch]
              + x[node * 3 + 1] * w[1 * HIDC + ch]
              + x[node * 3 + 2] * w[2 * HIDC + ch];
    h[idx] = acc;
}

// ---------------------------------------------------------------- fused per-edge message kernel
// Phase A (thread-per-edge): edge scalars -> z = silu(mlp1) (fp16 LDS); gather h[src],
//                            cross products -> c (fp16 LDS); dst (tail -> dummy row).
// Phase B (wave-per-16-edge-tile):
//   stage 1: gate GEMM  z(16x16,Kpad32) @ mlp2_w(16x48)  -> 3x v_wmma, bias+TP_NORM -> LDS
//   stage 2: channel mix c_k(16x16,Kpad32) @ tp_w(16x16) -> 3x v_wmma
// Phase C: branchless gated atomic scatter-add into h_nxt (L2-resident f32 atomics).
__global__ __launch_bounds__(TPB) void message_kernel(
    const float* __restrict__ h_cur, float* __restrict__ h_nxt,
    const int* __restrict__ edge_index, const float* __restrict__ edge_attr,
    const float* __restrict__ mlp1_w, const float* __restrict__ mlp1_b,
    const float* __restrict__ mlp2_w, const float* __restrict__ mlp2_b,
    const float* __restrict__ tp_w, int layer, int n_edges, int n_nodes)
{
    __shared__ float    gate_s[EPB][HIDC];       // 48 KB
    __shared__ _Float16 z_s[EPB][N_CHC];         //  8 KB
    __shared__ _Float16 c_s[EPB][3][N_CHC];      // 24 KB
    __shared__ int      dst_s[EPB];

    const int tid = threadIdx.x;
    const int e   = blockIdx.x * EPB + tid;

    const float* w1 = mlp1_w + layer * 3 * 16;
    const float* b1 = mlp1_b + layer * 16;
    const float* w2 = mlp2_w + layer * 16 * HIDC;
    const float* b2 = mlp2_b + layer * HIDC;

    // ---- Phase A: scalar per-edge work ----
    if (e < n_edges) {
        // warm L2 ahead (speculative prefetch, no counters)
        if (e + EPB * 64 < n_edges)
            __builtin_prefetch(&edge_attr[(size_t)(e + EPB * 64) * 5], 0, 1);

        float ea0 = edge_attr[(size_t)e * 5 + 0];
        float ea1 = edge_attr[(size_t)e * 5 + 1];
        float rx  = edge_attr[(size_t)e * 5 + 2];
        float ry  = edge_attr[(size_t)e * 5 + 3];
        float rz  = edge_attr[(size_t)e * 5 + 4];
        float r2  = rx * rx + ry * ry + rz * rz;
        float inv = 1.0f / fmaxf(sqrtf(r2), 1e-12f);
        const float s3 = 1.7320508075688772f;
        float shx = s3 * rx * inv, shy = s3 * ry * inv, shz = s3 * rz * inv;

        // z = silu([ea0,ea1,r2] @ w1 + b1)  (second MLP layer moves to WMMA in Phase B)
        #pragma unroll
        for (int j = 0; j < 16; ++j) {
            float zj = silu_f(ea0 * w1[0 * 16 + j] + ea1 * w1[1 * 16 + j]
                              + r2 * w1[2 * 16 + j] + b1[j]);
            z_s[tid][j] = (_Float16)zj;
        }

        int srcn   = edge_index[e];
        dst_s[tid] = edge_index[n_edges + e];

        // gather h[src] (48 floats, 192B-aligned) as 12x b128 loads; cross products -> fp16 LDS
        const float4* hp = (const float4*)(h_cur + (size_t)srcn * HIDC);
        #pragma unroll
        for (int u = 0; u < N_CHC; u += 4) {
            float4 p0 = hp[(u * 3) / 4 + 0];
            float4 p1 = hp[(u * 3) / 4 + 1];
            float4 p2 = hp[(u * 3) / 4 + 2];
            float xv[4] = {p0.x, p0.w, p1.z, p2.y};
            float yv[4] = {p0.y, p1.x, p1.w, p2.z};
            float zv[4] = {p0.z, p1.y, p2.x, p2.w};
            #pragma unroll
            for (int q = 0; q < 4; ++q) {
                int u2 = u + q;
                c_s[tid][0][u2] = (_Float16)(yv[q] * shz - zv[q] * shy);
                c_s[tid][1][u2] = (_Float16)(zv[q] * shx - xv[q] * shz);
                c_s[tid][2][u2] = (_Float16)(xv[q] * shy - yv[q] * shx);
            }
        }
    } else {
        dst_s[tid] = n_nodes;   // dummy trash row; message value is exactly 0 (c rows are 0)
        #pragma unroll
        for (int j = 0; j < 16; ++j) z_s[tid][j] = (_Float16)0.0f;
        #pragma unroll
        for (int u = 0; u < N_CHC; ++u) {
            c_s[tid][0][u] = (_Float16)0.0f;
            c_s[tid][1][u] = (_Float16)0.0f;
            c_s[tid][2][u] = (_Float16)0.0f;
        }
    }
    __syncthreads();

    // ---- Phase B/C setup ----
    const int lane = tid & 31;
    const int wave = tid >> 5;
    const int m    = lane & 15;
    const int hi   = lane >> 4;   // A: 0 -> K0..7 half, 1 -> K8..15 half; B rows 16..31 zero pad

    // B fragments (16-bit B layout: lane half hi, elem i<8 -> K=hi*8+i; elems 8..15 -> K=16..31 = 0)
    const float* tw = tp_w + layer * N_CHC * N_CHC;
    v16h b_tp, b_w2_0, b_w2_1, b_w2_2;
    #pragma unroll
    for (int i = 0; i < 16; ++i) {
        b_tp[i] = (_Float16)0.0f; b_w2_0[i] = (_Float16)0.0f;
        b_w2_1[i] = (_Float16)0.0f; b_w2_2[i] = (_Float16)0.0f;
    }
    {
        #pragma unroll
        for (int i = 0; i < 8; ++i) {
            b_tp[i]   = (_Float16)tw[(hi * 8 + i) * 16 + m];              // tp_w[k=hi*8+i][v=m]
            b_w2_0[i] = (_Float16)w2[(hi * 8 + i) * HIDC + 0  + m];       // mlp2_w[k=hi*8+i][o]
            b_w2_1[i] = (_Float16)w2[(hi * 8 + i) * HIDC + 16 + m];
            b_w2_2[i] = (_Float16)w2[(hi * 8 + i) * HIDC + 32 + m];
        }
    }
    // per-lane biases for the 3 gate column tiles
    float bias0 = b2[0 + m] * TP_NORM;
    float bias1 = b2[16 + m] * TP_NORM;
    float bias2 = b2[32 + m] * TP_NORM;

    #pragma unroll
    for (int t = 0; t < 2; ++t) {
        const int tile = wave * 2 + t;
        const int row0 = tile * 16;

        // ---- stage 1: gate GEMM (z @ mlp2_w), bias + TP_NORM folded, result -> gate_s ----
        v16h az;
        #pragma unroll
        for (int i = 0; i < 8; ++i)  az[i] = z_s[row0 + m][hi * 8 + i];
        #pragma unroll
        for (int i = 8; i < 16; ++i) az[i] = (_Float16)0.0f;

        v8f c0 = {}, c1 = {}, c2 = {};
        v8f g0 = __builtin_amdgcn_wmma_f32_16x16x32_f16(false, az, false, b_w2_0, (short)0, c0, false, false);
        v8f g1 = __builtin_amdgcn_wmma_f32_16x16x32_f16(false, az, false, b_w2_1, (short)0, c1, false, false);
        v8f g2 = __builtin_amdgcn_wmma_f32_16x16x32_f16(false, az, false, b_w2_2, (short)0, c2, false, false);
        #pragma unroll
        for (int r = 0; r < 8; ++r) {
            int row = r + hi * 8;
            gate_s[row0 + row][0  + m] = g0[r] * TP_NORM + bias0;
            gate_s[row0 + row][16 + m] = g1[r] * TP_NORM + bias1;
            gate_s[row0 + row][32 + m] = g2[r] * TP_NORM + bias2;
        }
        // same-wave LDS write->read is in-order; no barrier needed (tile owned by this wave)

        // hoist destination rows for this tile (branchless: tail edges -> dummy row)
        int dn[8];
        #pragma unroll
        for (int r = 0; r < 8; ++r) dn[r] = dst_s[row0 + r + hi * 8];

        // ---- stage 2: channel-mix GEMMs + branchless gated scatter ----
        #pragma unroll
        for (int k = 0; k < 3; ++k) {
            v16h ac;
            #pragma unroll
            for (int i = 0; i < 8; ++i)  ac[i] = c_s[row0 + m][k][hi * 8 + i];
            #pragma unroll
            for (int i = 8; i < 16; ++i) ac[i] = (_Float16)0.0f;

            v8f cc = {};
            v8f d = __builtin_amdgcn_wmma_f32_16x16x32_f16(false, ac, false, b_tp, (short)0, cc, false, false);

            #pragma unroll
            for (int r = 0; r < 8; ++r) {
                int eL    = row0 + r + hi * 8;
                float val = d[r] * gate_s[eL][m * 3 + k];   // TP_NORM already folded into gate
                atomicAdd(&h_nxt[(size_t)dn[r] * HIDC + m * 3 + k], val);
            }
        }
    }
}

// ---------------------------------------------------------------- pooling + readout
__global__ void zero_kernel(float* p, int n) {
    int i = blockIdx.x * blockDim.x + threadIdx.x;
    if (i < n) p[i] = 0.0f;
}

__global__ void pool_kernel(const float* __restrict__ h, const int* __restrict__ batch,
                            float* __restrict__ sums, float* __restrict__ cnt, int n_nodes) {
    int idx = blockIdx.x * blockDim.x + threadIdx.x;
    if (idx >= n_nodes * HIDC) return;
    int node = idx / HIDC, ch = idx % HIDC;
    int g = batch[node];
    atomicAdd(&sums[g * HIDC + ch], h[idx]);
    if (ch == 0) atomicAdd(&cnt[g], 1.0f);
}

__global__ void readout_kernel(const float* __restrict__ sums, const float* __restrict__ cnt,
                               const float* __restrict__ lin_w, const float* __restrict__ lin_b,
                               float* __restrict__ out) {
    int g = blockIdx.x * blockDim.x + threadIdx.x;
    if (g >= N_GRAPHSC) return;
    float inv = 1.0f / fmaxf(cnt[g], 1.0f);
    float acc = lin_b[0];
    #pragma unroll
    for (int c = 0; c < HIDC; ++c) acc += sums[g * HIDC + c] * inv * lin_w[c];
    out[g] = acc;
}

// ---------------------------------------------------------------- launcher
extern "C" void kernel_launch(void* const* d_in, const int* in_sizes, int n_in,
                              void* d_out, int out_size, void* d_ws, size_t ws_size,
                              hipStream_t stream) {
    (void)n_in; (void)out_size; (void)ws_size;
    const float* x          = (const float*)d_in[0];
    const int*   edge_index = (const int*)  d_in[1];
    const float* edge_attr  = (const float*)d_in[2];
    const int*   batch      = (const int*)  d_in[3];
    const float* embed_w    = (const float*)d_in[4];
    const float* mlp1_w     = (const float*)d_in[5];
    const float* mlp1_b     = (const float*)d_in[6];
    const float* mlp2_w     = (const float*)d_in[7];
    const float* mlp2_b     = (const float*)d_in[8];
    const float* tp_w       = (const float*)d_in[9];
    const float* lin_w      = (const float*)d_in[10];
    const float* lin_b      = (const float*)d_in[11];

    const int n_nodes = in_sizes[0] / 3;
    const int n_edges = in_sizes[1] / 2;

    // h buffers have one extra trash row (dummy scatter target for tail edges)
    const size_t hrows = (size_t)n_nodes + 1;
    float* h_a  = (float*)d_ws;
    float* h_b  = h_a + hrows * HIDC;
    float* sums = h_b + hrows * HIDC;             // [64][48] then [64] cnt, contiguous
    float* cnt  = sums + N_GRAPHSC * HIDC;

    embed_kernel<<<(n_nodes * HIDC + TPB - 1) / TPB, TPB, 0, stream>>>(x, embed_w, h_a, n_nodes);

    float* cur = h_a;
    float* nxt = h_b;
    for (int l = 0; l < N_LAYERSC; ++l) {
        hipMemcpyAsync(nxt, cur, (size_t)n_nodes * HIDC * sizeof(float),
                       hipMemcpyDeviceToDevice, stream);
        message_kernel<<<(n_edges + EPB - 1) / EPB, TPB, 0, stream>>>(
            cur, nxt, edge_index, edge_attr,
            mlp1_w, mlp1_b, mlp2_w, mlp2_b, tp_w, l, n_edges, n_nodes);
        float* tmp = cur; cur = nxt; nxt = tmp;
    }

    zero_kernel<<<(N_GRAPHSC * HIDC + N_GRAPHSC + TPB - 1) / TPB, TPB, 0, stream>>>(
        sums, N_GRAPHSC * HIDC + N_GRAPHSC);
    pool_kernel<<<(n_nodes * HIDC + TPB - 1) / TPB, TPB, 0, stream>>>(cur, batch, sums, cnt, n_nodes);
    readout_kernel<<<1, N_GRAPHSC, 0, stream>>>(sums, cnt, lin_w, lin_b, (float*)d_out);
}
